// AGCN_Block_76424648065376
// MI455X (gfx1250) — matrix operand order repaired and checked by reading, not compiled, not run on hardware
//
#include <hip/hip_runtime.h>
#include <hip/hip_bf16.h>

#define BB 64
#define NN 256
#define TT 512
#define KORD 3
#define NMAX 51          // NN / 5
#define NT (NN * TT)     // 131072
#define BNT (BB * NN * TT)
#define BNN (BB * NN * NN)

typedef _Float16 h16;
typedef __attribute__((ext_vector_type(16))) _Float16 v16h;
typedef __attribute__((ext_vector_type(8)))  _Float16 v8h;
typedef __attribute__((ext_vector_type(8)))  float    v8f;

// ---------------- WMMA fragment helpers (wave32, 16x16x32 f16->f32) ---------

__device__ __forceinline__ v8f wmma_f16(v16h a, v16h b, v8f c) {
  return __builtin_amdgcn_wmma_f32_16x16x32_f16(false, a, false, b, (short)0, c,
                                                false, false);
}

// A fragment, 16x32 f16, row-major source (lda in elements).
__device__ __forceinline__ v16h load_a_f16(const h16* A, int lda, int lane) {
  const int m  = lane & 15;
  const int kb = (lane < 16) ? 0 : 8;
  const h16* p = A + (size_t)m * lda + kb;
  v8h lo = *(const v8h*)(p);        // K = kb .. kb+7
  v8h hi = *(const v8h*)(p + 16);   // K = 16+kb .. 16+kb+7
  return __builtin_shufflevector(lo, hi, 0, 1, 2, 3, 4, 5, 6, 7,
                                 8, 9, 10, 11, 12, 13, 14, 15);
}

// A fragment from fp32 source (convert on the fly; compiles to b128 + cvt_pk).
__device__ __forceinline__ v16h load_a_f32(const float* A, int lda, int lane) {
  const int m  = lane & 15;
  const int kb = (lane < 16) ? 0 : 8;
  const float* p = A + (size_t)m * lda + kb;
  v16h r;
#pragma unroll
  for (int i = 0; i < 8; ++i) {
    r[i]     = (_Float16)p[i];
    r[8 + i] = (_Float16)p[16 + i];
  }
  return r;
}

// B fragment: operand supplied K-contiguous (row-major Ncols x K).
// Two aligned 16B v8h loads per lane (K = kb..kb+7, 16+kb..16+kb+7).
__device__ __forceinline__ v16h load_bt_f16(const h16* Bt, int ldb, int lane) {
  const int n  = lane & 15;
  const int kb = (lane < 16) ? 0 : 8;
  const h16* p = Bt + (size_t)n * ldb + kb;
  v8h lo = *(const v8h*)(p);
  v8h hi = *(const v8h*)(p + 16);
  return __builtin_shufflevector(lo, hi, 0, 1, 2, 3, 4, 5, 6, 7,
                                 8, 9, 10, 11, 12, 13, 14, 15);
}

// ---------------- 1) LayerNorm over (N,T) per batch -------------------------

__global__ void ln_kernel(const float* __restrict__ x,
                          const float* __restrict__ ln_w,
                          const float* __restrict__ ln_b,
                          float* __restrict__ x_ln) {
  const int b = blockIdx.x;
  const int tid = threadIdx.x;
  const float* xb = x + (size_t)b * NT;
  float s = 0.f, ss = 0.f;
  for (int i = tid; i < NT; i += 256) {
    float v = xb[i];
    s += v;
    ss += v * v;
  }
  __shared__ float rs[256], rq[256];
  rs[tid] = s; rq[tid] = ss;
  __syncthreads();
  for (int off = 128; off > 0; off >>= 1) {
    if (tid < off) { rs[tid] += rs[tid + off]; rq[tid] += rq[tid + off]; }
    __syncthreads();
  }
  __shared__ float s_mu, s_rstd;
  if (tid == 0) {
    float mu  = rs[0] * (1.0f / NT);
    float var = rq[0] * (1.0f / NT) - mu * mu;
    s_mu = mu;
    s_rstd = rsqrtf(var + 1e-5f);
  }
  __syncthreads();
  const float mu = s_mu, rstd = s_rstd;
  float* ob = x_ln + (size_t)b * NT;
  for (int i = tid; i < NT; i += 256)
    ob[i] = (xb[i] - mu) * rstd * ln_w[i] + ln_b[i];
}

// ---------------- fp32 -> fp16 copy / transpose-copy ------------------------

__global__ void cvt_kernel(const float* __restrict__ src, h16* __restrict__ dst,
                           int n) {
  int i = blockIdx.x * 256 + threadIdx.x;
  if (i < n) dst[i] = (_Float16)src[i];
}

// cheb_w (K,T,T)[k][t][f] -> dst[k][f][t]  (K-contiguous for B fragments)
__global__ void cvt_chebT_kernel(const float* __restrict__ src,
                                 h16* __restrict__ dst) {
  int i = blockIdx.x * 256 + threadIdx.x;     // over KORD*TT*TT
  int k = i / (TT * TT);
  int r = i % (TT * TT);
  int t = r / TT;
  int f = r % TT;
  dst[(size_t)k * TT * TT + (size_t)f * TT + t] = (_Float16)src[i];
}

// ---------------- 2) xp = x_ln @ li_w^T + li_b  (WMMA, 16x32 tile/wave) -----

__global__ void gemm_xp_kernel(const float* __restrict__ x_ln,
                               const h16* __restrict__ liw_h,
                               const float* __restrict__ li_b,
                               h16* __restrict__ xp_h) {
  const int lane = threadIdx.x & 31;
  const int wave = (blockIdx.x * blockDim.x + threadIdx.x) >> 5;
  const int TPB  = (NN / 16) * (TT / 32);   // 16 * 16 = 256 tiles/batch
  const int b    = wave / TPB;
  const int t_   = wave % TPB;
  const int tm   = t_ / (TT / 32);
  const int tp   = t_ % (TT / 32);          // pair of 16-col groups
  const float* A  = x_ln + (size_t)b * NT + (size_t)(tm * 16) * TT;
  const h16* Bt0  = liw_h + (size_t)(tp * 32) * TT;
  const h16* Bt1  = Bt0 + (size_t)16 * TT;
  v8f c0 = {}, c1 = {};
  for (int k0 = 0; k0 < TT; k0 += 32) {
    v16h a = load_a_f32(A + k0, TT, lane);
    c0 = wmma_f16(a, load_bt_f16(Bt0 + k0, TT, lane), c0);
    c1 = wmma_f16(a, load_bt_f16(Bt1 + k0, TT, lane), c1);
  }
  const int col = lane & 15;
  const int rb  = (lane < 16) ? 0 : 8;
  const int f0  = tp * 32 + col;
  const float b0 = li_b[f0], b1 = li_b[f0 + 16];
  h16* D = xp_h + (size_t)b * NT + (size_t)(tm * 16) * TT + f0;
#pragma unroll
  for (int r = 0; r < 8; ++r) {
    D[(size_t)(rb + r) * TT]      = (_Float16)(c0[r] + b0);
    D[(size_t)(rb + r) * TT + 16] = (_Float16)(c1[r] + b1);
  }
}

// ---------------- 3) scores = xp @ xp^T per batch (WMMA) --------------------

__global__ void gemm_scores_kernel(const h16* __restrict__ xp_h,
                                   float* __restrict__ scores) {
  const int lane = threadIdx.x & 31;
  const int wave = (blockIdx.x * blockDim.x + threadIdx.x) >> 5;
  const int TPB  = (NN / 16) * (NN / 32);   // 16 * 8 = 128 tiles/batch
  const int b    = wave / TPB;
  const int t_   = wave % TPB;
  const int tm   = t_ / (NN / 32);
  const int tp   = t_ % (NN / 32);
  const h16* A   = xp_h + (size_t)b * NT + (size_t)(tm * 16) * TT;
  const h16* Bt0 = xp_h + (size_t)b * NT + (size_t)(tp * 32) * TT;
  const h16* Bt1 = Bt0 + (size_t)16 * TT;
  v8f c0 = {}, c1 = {};
  for (int k0 = 0; k0 < TT; k0 += 32) {
    v16h a = load_a_f16(A + k0, TT, lane);
    c0 = wmma_f16(a, load_bt_f16(Bt0 + k0, TT, lane), c0);
    c1 = wmma_f16(a, load_bt_f16(Bt1 + k0, TT, lane), c1);
  }
  const int col = lane & 15;
  const int rb  = (lane < 16) ? 0 : 8;
  float* D = scores + (size_t)b * NN * NN + (size_t)(tm * 16) * NN + tp * 32 + col;
#pragma unroll
  for (int r = 0; r < 8; ++r) {
    D[(size_t)(rb + r) * NN]      = c0[r];
    D[(size_t)(rb + r) * NN + 16] = c1[r];
  }
}

// ---------------- 4) top-k sparsify + dis_adj + relu + row degree -----------

__global__ void topk_kernel(float* __restrict__ scores,
                            const float* __restrict__ dis_adj,
                            float* __restrict__ deg) {
  const int row = blockIdx.x;          // b*N + n
  const int n   = row & (NN - 1);
  const int tid = threadIdx.x;         // 256 == NN
  float* S = scores + (size_t)row * NN;
  __shared__ float sv[NN];
  float v = S[tid];
  sv[tid] = v;
  __syncthreads();
  int gt = 0, eq = 0;
#pragma unroll 8
  for (int j = 0; j < NN; ++j) {
    float u = sv[j];
    gt += (u > v);
    eq += (u == v);
  }
  __shared__ float s_kth;
  // v is the NMAX-th largest value iff fewer than NMAX are strictly greater
  // and at least NMAX are >= v (duplicate-safe: all matches share one value).
  if (gt < NMAX && gt + eq >= NMAX) s_kth = v;
  __syncthreads();
  const float kth = s_kth;
  float a = (v >= kth) ? v : 0.f;
  a = fmaxf(a + dis_adj[(size_t)n * NN + tid], 0.f);
  S[tid] = a;
  __shared__ float rs[NN];
  rs[tid] = a;
  __syncthreads();
  for (int off = 128; off > 0; off >>= 1) {
    if (tid < off) rs[tid] += rs[tid + off];
    __syncthreads();
  }
  if (tid == 0) deg[row] = rs[0];
}

// ---------------- 5) L = D^-1/2 A D^-1/2 (to f16) ---------------------------

__global__ void L_kernel(const float* __restrict__ A,
                         const float* __restrict__ deg,
                         h16* __restrict__ L_h) {
  const int idx = blockIdx.x * 256 + threadIdx.x;   // over B*N*N
  const int m   = idx & (NN - 1);
  const int rn  = idx >> 8;           // b*N + n
  const int b   = rn >> 8;
  const float di = rsqrtf(deg[rn] + 1e-10f);
  const float dj = rsqrtf(deg[b * NN + m] + 1e-10f);
  L_h[idx] = (_Float16)(di * A[idx] * dj);
}

// ---------------- 6) BatchNorm stats / apply over (B,T) per node ------------

__global__ void bn_stats_kernel(const float* __restrict__ x_ln,
                                float* __restrict__ bn_mean,
                                float* __restrict__ bn_rstd) {
  const int n = blockIdx.x;
  const int tid = threadIdx.x;
  float s = 0.f, ss = 0.f;
  for (int i = tid; i < BB * TT; i += 256) {
    int b = i / TT, t = i % TT;
    float v = x_ln[((size_t)b * NN + n) * TT + t];
    s += v;
    ss += v * v;
  }
  __shared__ float rs[256], rq[256];
  rs[tid] = s; rq[tid] = ss;
  __syncthreads();
  for (int off = 128; off > 0; off >>= 1) {
    if (tid < off) { rs[tid] += rs[tid + off]; rq[tid] += rq[tid + off]; }
    __syncthreads();
  }
  if (tid == 0) {
    const float inv = 1.0f / (BB * TT);
    float mu  = rs[0] * inv;
    float var = rq[0] * inv - mu * mu;
    bn_mean[n] = mu;
    bn_rstd[n] = rsqrtf(var + 1e-5f);
  }
}

// Writes x_bn (f32, N x T) and a transposed f16 copy x_bn_t (T x N) so the
// T1 GEMM can load its B fragments K-contiguously (two v8h loads per lane).
__global__ void bn_apply_kernel(const float* __restrict__ x_ln,
                                const float* __restrict__ bn_mean,
                                const float* __restrict__ bn_rstd,
                                const float* __restrict__ bn_g,
                                const float* __restrict__ bn_b,
                                float* __restrict__ x_bn,
                                h16* __restrict__ x_bn_t) {
  const int idx = blockIdx.x * 256 + threadIdx.x;   // over B*N*T
  const int t = idx & (TT - 1);
  const int n = (idx / TT) & (NN - 1);
  const int b = idx / NT;
  float v = (x_ln[idx] - bn_mean[n]) * bn_rstd[n] * bn_g[n] + bn_b[n];
  x_bn[idx] = v;
  x_bn_t[(size_t)b * NT + (size_t)t * NN + n] = (_Float16)v;
}

// ---------------- 7) Tx1 = L @ Tx0 (WMMA) -----------------------------------
// Stores Tx1 row-major (A-operand for cheb) and transposed (B-operand for T2).
// Transposed store is contiguous per lane: one v8h (b128) store per tile row.

__global__ void gemm_T1_kernel(const h16* __restrict__ L_h,
                               const h16* __restrict__ x_bn_t,
                               h16* __restrict__ tx1_h,
                               h16* __restrict__ tx1_t) {
  const int lane = threadIdx.x & 31;
  const int wave = (blockIdx.x * blockDim.x + threadIdx.x) >> 5;
  const int TPB  = (NN / 16) * (TT / 32);   // 256 tiles/batch
  const int b    = wave / TPB;
  const int t_   = wave % TPB;
  const int tm   = t_ / (TT / 32);
  const int tp   = t_ % (TT / 32);
  const h16* A   = L_h + (size_t)b * NN * NN + (size_t)(tm * 16) * NN;
  const h16* Bt0 = x_bn_t + (size_t)b * NT + (size_t)(tp * 32) * NN;
  const h16* Bt1 = Bt0 + (size_t)16 * NN;
  v8f c0 = {}, c1 = {};
  for (int k0 = 0; k0 < NN; k0 += 32) {
    v16h a = load_a_f16(A + k0, NN, lane);
    c0 = wmma_f16(a, load_bt_f16(Bt0 + k0, NN, lane), c0);
    c1 = wmma_f16(a, load_bt_f16(Bt1 + k0, NN, lane), c1);
  }
  const int col = lane & 15;
  const int rb  = (lane < 16) ? 0 : 8;
  v8h p0, p1;
#pragma unroll
  for (int r = 0; r < 8; ++r) { p0[r] = (_Float16)c0[r]; p1[r] = (_Float16)c1[r]; }
  // row-major copy (strided scalar stores)
  h16* D = tx1_h + (size_t)b * NT + (size_t)(tm * 16) * TT + tp * 32 + col;
#pragma unroll
  for (int r = 0; r < 8; ++r) {
    D[(size_t)(rb + r) * TT]      = p0[r];
    D[(size_t)(rb + r) * TT + 16] = p1[r];
  }
  // transposed copy (contiguous b128 stores)
  h16* Dt = tx1_t + (size_t)b * NT + (size_t)(tp * 32 + col) * NN + tm * 16 + rb;
  *(v8h*)(Dt) = p0;
  *(v8h*)(Dt + (size_t)16 * NN) = p1;
}

// ---------------- 8) Tx2 = 2 * (L @ Tx1) - Tx0 (WMMA) -----------------------

__global__ void gemm_T2_kernel(const h16* __restrict__ L_h,
                               const h16* __restrict__ tx1_t,
                               const float* __restrict__ x_bn,
                               h16* __restrict__ tx2_h) {
  const int lane = threadIdx.x & 31;
  const int wave = (blockIdx.x * blockDim.x + threadIdx.x) >> 5;
  const int TPB  = (NN / 16) * (TT / 32);
  const int b    = wave / TPB;
  const int t_   = wave % TPB;
  const int tm   = t_ / (TT / 32);
  const int tp   = t_ % (TT / 32);
  const h16* A   = L_h + (size_t)b * NN * NN + (size_t)(tm * 16) * NN;
  const h16* Bt0 = tx1_t + (size_t)b * NT + (size_t)(tp * 32) * NN;
  const h16* Bt1 = Bt0 + (size_t)16 * NN;
  v8f c0 = {}, c1 = {};
  for (int k0 = 0; k0 < NN; k0 += 32) {
    v16h a = load_a_f16(A + k0, NN, lane);
    c0 = wmma_f16(a, load_bt_f16(Bt0 + k0, NN, lane), c0);
    c1 = wmma_f16(a, load_bt_f16(Bt1 + k0, NN, lane), c1);
  }
  const int col = lane & 15;
  const int rb  = (lane < 16) ? 0 : 8;
  const size_t base = (size_t)b * NT + (size_t)(tm * 16) * TT + tp * 32 + col;
#pragma unroll
  for (int r = 0; r < 8; ++r) {
    size_t o = base + (size_t)(rb + r) * TT;
    tx2_h[o]      = (_Float16)(2.0f * c0[r] - x_bn[o]);
    tx2_h[o + 16] = (_Float16)(2.0f * c1[r] - x_bn[o + 16]);
  }
}

// ---------------- 9) out = relu(sum_k Tx_k @ W_k + cheb_b) (fused WMMA) -----

__global__ void gemm_cheb_kernel(const float* __restrict__ x_bn,
                                 const h16* __restrict__ tx1_h,
                                 const h16* __restrict__ tx2_h,
                                 const h16* __restrict__ chebw_t,
                                 const float* __restrict__ cheb_b,
                                 float* __restrict__ out) {
  const int lane = threadIdx.x & 31;
  const int wave = (blockIdx.x * blockDim.x + threadIdx.x) >> 5;
  const int TPB  = (NN / 16) * (TT / 32);
  const int b    = wave / TPB;
  const int t_   = wave % TPB;
  const int tm   = t_ / (TT / 32);
  const int tp   = t_ % (TT / 32);
  const size_t arow = (size_t)b * NT + (size_t)(tm * 16) * TT;
  v8f c0 = {}, c1 = {};
  // k = 0: Tx0 = x_bn (fp32 source, convert in A loader)
  {
    const float* A = x_bn + arow;
    const h16* W0  = chebw_t + (size_t)(tp * 32) * TT;
    const h16* W1  = W0 + (size_t)16 * TT;
    for (int t0 = 0; t0 < TT; t0 += 32) {
      v16h a = load_a_f32(A + t0, TT, lane);
      c0 = wmma_f16(a, load_bt_f16(W0 + t0, TT, lane), c0);
      c1 = wmma_f16(a, load_bt_f16(W1 + t0, TT, lane), c1);
    }
  }
  // k = 1
  {
    const h16* A  = tx1_h + arow;
    const h16* W0 = chebw_t + (size_t)TT * TT + (size_t)(tp * 32) * TT;
    const h16* W1 = W0 + (size_t)16 * TT;
    for (int t0 = 0; t0 < TT; t0 += 32) {
      v16h a = load_a_f16(A + t0, TT, lane);
      c0 = wmma_f16(a, load_bt_f16(W0 + t0, TT, lane), c0);
      c1 = wmma_f16(a, load_bt_f16(W1 + t0, TT, lane), c1);
    }
  }
  // k = 2
  {
    const h16* A  = tx2_h + arow;
    const h16* W0 = chebw_t + (size_t)2 * TT * TT + (size_t)(tp * 32) * TT;
    const h16* W1 = W0 + (size_t)16 * TT;
    for (int t0 = 0; t0 < TT; t0 += 32) {
      v16h a = load_a_f16(A + t0, TT, lane);
      c0 = wmma_f16(a, load_bt_f16(W0 + t0, TT, lane), c0);
      c1 = wmma_f16(a, load_bt_f16(W1 + t0, TT, lane), c1);
    }
  }
  const int col = lane & 15;
  const int rb  = (lane < 16) ? 0 : 8;
  const int f0  = tp * 32 + col;
  const float b0 = cheb_b[f0], b1 = cheb_b[f0 + 16];
  float* D = out + arow + f0;
#pragma unroll
  for (int r = 0; r < 8; ++r) {
    D[(size_t)(rb + r) * TT]      = fmaxf(c0[r] + b0, 0.f);
    D[(size_t)(rb + r) * TT + 16] = fmaxf(c1[r] + b1, 0.f);
  }
}

// ---------------- host launch ------------------------------------------------

extern "C" void kernel_launch(void* const* d_in, const int* in_sizes, int n_in,
                              void* d_out, int out_size, void* d_ws,
                              size_t ws_size, hipStream_t stream) {
  const float* x       = (const float*)d_in[0];
  const float* dis_adj = (const float*)d_in[1];
  const float* ln_w    = (const float*)d_in[2];
  const float* ln_b    = (const float*)d_in[3];
  const float* bn_g    = (const float*)d_in[4];
  const float* bn_b    = (const float*)d_in[5];
  const float* li_w    = (const float*)d_in[6];
  const float* li_b    = (const float*)d_in[7];
  const float* cheb_w  = (const float*)d_in[8];
  const float* cheb_b  = (const float*)d_in[9];
  float* out = (float*)d_out;

  // workspace carve-out (offsets kept 256B-aligned)
  char* ws = (char*)d_ws;
  size_t off = 0;
  auto take = [&](size_t bytes) {
    char* p = ws + off;
    off += (bytes + 255) & ~(size_t)255;
    return p;
  };
  float* x_ln     = (float*)take((size_t)BNT * 4);
  h16*   xp_h     = (h16*)  take((size_t)BNT * 2);
  h16*   liw_h    = (h16*)  take((size_t)TT * TT * 2);
  h16*   chebw_t  = (h16*)  take((size_t)KORD * TT * TT * 2);
  float* scores   = (float*)take((size_t)BNN * 4);
  float* deg      = (float*)take((size_t)BB * NN * 4);
  h16*   L_h      = (h16*)  take((size_t)BNN * 2);
  float* bn_mean  = (float*)take((size_t)NN * 4);
  float* bn_rstd  = (float*)take((size_t)NN * 4);
  float* x_bn     = (float*)take((size_t)BNT * 4);
  h16*   x_bn_t   = (h16*)  take((size_t)BNT * 2);
  h16*   tx1_h    = (h16*)  take((size_t)BNT * 2);
  h16*   tx1_t    = (h16*)  take((size_t)BNT * 2);
  h16*   tx2_h    = (h16*)  take((size_t)BNT * 2);
  (void)ws_size; (void)in_sizes; (void)n_in; (void)out_size;

  const int blk_NT = (BB * (NN / 16) * (TT / 32)) / 8;   // 2048 blocks
  const int blk_NN = (BB * (NN / 16) * (NN / 32)) / 8;   // 1024 blocks

  ln_kernel<<<BB, 256, 0, stream>>>(x, ln_w, ln_b, x_ln);
  cvt_kernel<<<(TT * TT + 255) / 256, 256, 0, stream>>>(li_w, liw_h, TT * TT);
  cvt_chebT_kernel<<<(KORD * TT * TT + 255) / 256, 256, 0, stream>>>(cheb_w,
                                                                     chebw_t);
  gemm_xp_kernel<<<blk_NT, 256, 0, stream>>>(x_ln, liw_h, li_b, xp_h);
  gemm_scores_kernel<<<blk_NN, 256, 0, stream>>>(xp_h, scores);
  topk_kernel<<<BB * NN, 256, 0, stream>>>(scores, dis_adj, deg);
  L_kernel<<<BNN / 256, 256, 0, stream>>>(scores, deg, L_h);
  bn_stats_kernel<<<NN, 256, 0, stream>>>(x_ln, bn_mean, bn_rstd);
  bn_apply_kernel<<<BNT / 256, 256, 0, stream>>>(x_ln, bn_mean, bn_rstd, bn_g,
                                                 bn_b, x_bn, x_bn_t);
  gemm_T1_kernel<<<blk_NT, 256, 0, stream>>>(L_h, x_bn_t, tx1_h, tx1_t);
  gemm_T2_kernel<<<blk_NT, 256, 0, stream>>>(L_h, tx1_t, x_bn, tx2_h);
  gemm_cheb_kernel<<<blk_NT, 256, 0, stream>>>(x_bn, tx1_h, tx2_h, chebw_t,
                                               cheb_b, out);
}
